// SpatialGeomEncoder_81329500717512
// MI455X (gfx1250) — matrix operand order repaired and checked by reading, not compiled, not run on hardware
//
#include <hip/hip_runtime.h>

// ---------------------------------------------------------------------------
// Types for CDNA5 WMMA (wave32, v_wmma_f32_16x16x32_bf16) and TDM descriptors
// ---------------------------------------------------------------------------
typedef __bf16 bf16;
typedef __bf16 v16bf __attribute__((ext_vector_type(16)));
typedef float  v8f   __attribute__((ext_vector_type(8)));
typedef unsigned int u32x4 __attribute__((ext_vector_type(4)));
typedef int          i32x4 __attribute__((ext_vector_type(4)));
typedef int          i32x8 __attribute__((ext_vector_type(8)));

#define NEG_SLOPE 0.2f
#define TWO_PI    6.28318530717958647692f
#define LN32      3.46573590279972654709f

#if defined(__HIP_DEVICE_COMPILE__) && __has_builtin(__builtin_amdgcn_tensor_load_to_lds) && __has_builtin(__builtin_amdgcn_s_wait_tensorcnt)
#define HAVE_TDM 1
#else
#define HAVE_TDM 0
#endif

static inline int cdiv(long long a, long long b) { return (int)((a + b - 1) / b); }

// ---------------------------------------------------------------------------
// Fragment loaders (layouts per CDNA5 ISA §7.12.2, wave32)
// A: 16-bit 16x32. lane m (0-15): v0-3 = K 0..7, v4-7 = K 16..23
//                  lane m+16   : v0-3 = K 8..15, v4-7 = K 24..31
// ---------------------------------------------------------------------------
__device__ __forceinline__ v16bf load_a_frag(const bf16* base, int row0, int ld,
                                             int kc, int lane) {
  int m  = lane & 15;
  int kg = lane >> 4;
  const bf16* p = base + (size_t)(row0 + m) * ld + kc * 32 + kg * 8;
  union { v16bf v; uint4 u[2]; } a;
  a.u[0] = *(const uint4*)(p);
  a.u[1] = *(const uint4*)(p + 16);
  return a.v;
}

// B fragments are pre-packed so each lane loads 16 contiguous bf16 (32B).
// Works for both global and LDS (generic) pointers.
__device__ __forceinline__ v16bf load_b_frag(const bf16* packed, int nb, int Kc,
                                             int kc, int lane) {
  const bf16* p = packed + ((size_t)(nb * Kc + kc) * 32 + lane) * 16;
  union { v16bf v; uint4 u[2]; } b;
  b.u[0] = *(const uint4*)(p);
  b.u[1] = *(const uint4*)(p + 8);
  return b.v;
}

__device__ __forceinline__ v8f wmma_bf16(v16bf a, v16bf b, v8f c) {
  return __builtin_amdgcn_wmma_f32_16x16x32_bf16(false, a, false, b, (short)0, c,
                                                 false, false);
}

__device__ __forceinline__ void atomicMaxFloat(float* addr, float val) {
  if (val >= 0.0f) atomicMax((int*)addr, __float_as_int(val));
  else             atomicMin((unsigned int*)addr, __float_as_uint(val));
}

__device__ __forceinline__ float silu(float x) { return x / (1.0f + __expf(-x)); }

// ---------------------------------------------------------------------------
// TDM: stage n_dwords (32-bit words) from global to LDS with one descriptor.
// D# built per CDNA5 ISA: 2-D tensor, data_size=4B,
// tensor_dim0 = tile_dim0 = n_dwords, dim1 = 1. Issued by one wave, EXEC
// ignored by TDM; caller must s_wait_tensorcnt + barrier.
// ---------------------------------------------------------------------------
#if HAVE_TDM
__device__ __forceinline__ void tdm_load_to_lds(const void* gaddr, void* lds_ptr,
                                                int n_dwords) {
  unsigned long long ga = (unsigned long long)(uintptr_t)gaddr;
  u32x4 g0 = {};
  g0[0] = 1u;                                          // count=1 (valid), user mode
  g0[1] = (unsigned)(uintptr_t)lds_ptr;                // lds_addr (LDS byte offset)
  g0[2] = (unsigned)(ga & 0xFFFFFFFFu);                // global_addr[31:0]
  g0[3] = (unsigned)((ga >> 32) & 0x01FFFFFFu) | (2u << 30);  // addr[56:32] | type=2
  i32x8 g1 = {};
  g1[0] = (2 << 16);                                   // data_size = 4 bytes
  g1[1] = (n_dwords & 0xFFFF) << 16;                   // tensor_dim0[15:0] @bit48
  g1[2] = ((n_dwords >> 16) & 0xFFFF) | (1 << 16);     // tensor_dim0[31:16], dim1=1
  g1[3] = (n_dwords & 0xFFFF) << 16;                   // tile_dim0 @bit112
  g1[4] = 1;                                           // tile_dim1 = 1
  g1[5] = n_dwords;                                    // tensor_dim0_stride[31:0]
  i32x4 z4 = {};
#if __clang_major__ >= 23
  i32x8 z8 = {};
  __builtin_amdgcn_tensor_load_to_lds(g0, g1, z4, z4, z8, 0);
#else
  __builtin_amdgcn_tensor_load_to_lds(g0, g1, z4, z4, 0);
#endif
}
#endif

// Cooperative block-wide staging: TDM when available, else vector copy.
__device__ __forceinline__ void stage_to_lds(const bf16* g, bf16* s, int n_bytes) {
#if HAVE_TDM
  if ((threadIdx.x >> 5) == 0) {
    tdm_load_to_lds(g, s, n_bytes >> 2);
    __builtin_amdgcn_s_wait_tensorcnt(0);
  }
#else
  const uint4* g4 = (const uint4*)g;
  uint4* s4 = (uint4*)s;
  int n16 = n_bytes >> 4;
  for (int i = threadIdx.x; i < n16; i += blockDim.x) s4[i] = g4[i];
#endif
  __syncthreads();
}

// ---------------------------------------------------------------------------
// Kernel: node positional encoding -> h_pad bf16 (N x 128, cols 44..127 = 0)
// ---------------------------------------------------------------------------
__global__ void pe_kernel(const float* __restrict__ x, bf16* __restrict__ h_pad, int N) {
  int n = blockIdx.x * blockDim.x + threadIdx.x;
  if (n >= N) return;
  const float* xr = x + (size_t)n * 12;
  bf16* hr = h_pad + (size_t)n * 128;
#pragma unroll
  for (int k = 0; k < 12; ++k) hr[k] = (bf16)xr[k];
  float r = 1.0f / (1.0f + __expf(-xr[1]));
  float c = 1.0f / (1.0f + __expf(-xr[2]));
#pragma unroll
  for (int i = 0; i < 8; ++i) {
    float f = __expf(LN32 * (float)i / 7.0f);
    float ar = r * f * TWO_PI;
    float ac = c * f * TWO_PI;
    hr[12 + i]      = (bf16)__sinf(ar);
    hr[12 + 8 + i]  = (bf16)__cosf(ar);
    hr[28 + i]      = (bf16)__sinf(ac);
    hr[28 + 8 + i]  = (bf16)__cosf(ac);
  }
#pragma unroll
  for (int k = 44; k < 128; ++k) hr[k] = (bf16)0.0f;
}

// ---------------------------------------------------------------------------
// Kernel: zero float buffer
// ---------------------------------------------------------------------------
__global__ void zero_kernel(float* __restrict__ p, int n) {
  int t = blockIdx.x * blockDim.x + threadIdx.x;
  if (t < n) p[t] = 0.0f;
}

// ---------------------------------------------------------------------------
// Kernel: edge MLP  e = silu(edge_attr @ W + b), bf16 out; deg/e_sum atomics
// one wave per edge; lane handles 4 channels; edge_attr loaded once per wave
// and broadcast via lane shuffles.
// ---------------------------------------------------------------------------
__global__ void edge_mlp_kernel(const float* __restrict__ edge_attr,
                                const float* __restrict__ edge_w,
                                const float* __restrict__ edge_b,
                                const int*   __restrict__ dst_idx,
                                bf16*  __restrict__ e_f,
                                float* __restrict__ e_sum,
                                float* __restrict__ deg, int E) {
  int t = blockIdx.x * blockDim.x + threadIdx.x;
  int e = t >> 5, l = t & 31;
  if (e >= E) return;
  float mine = (l < 9) ? edge_attr[(size_t)e * 9 + l] : 0.0f;
  float ea[9];
#pragma unroll
  for (int k = 0; k < 9; ++k) ea[k] = __shfl(mine, k, 32);
  int d = dst_idx[e];
  if (l == 0) atomicAdd(&deg[d], 1.0f);
  int c0 = l * 4;
  bf16* dst = e_f + (size_t)e * 128 + c0;
#pragma unroll
  for (int j = 0; j < 4; ++j) {
    int c = c0 + j;
    float acc = edge_b[c];
#pragma unroll
    for (int k = 0; k < 9; ++k) acc += ea[k] * edge_w[k * 128 + c];
    acc = silu(acc);
    dst[j] = (bf16)acc;
    atomicAdd(&e_sum[(size_t)d * 128 + c], acc);
  }
}

// ---------------------------------------------------------------------------
// Kernel: self-loop edge features = mean of incoming -> e_f rows [E, E+N)
// ---------------------------------------------------------------------------
__global__ void emean_kernel(const float* __restrict__ e_sum,
                             const float* __restrict__ deg,
                             bf16* __restrict__ e_f, int N, int E) {
  int t = blockIdx.x * blockDim.x + threadIdx.x;
  int n = t >> 5, l = t & 31;
  if (n >= N) return;
  float inv = 1.0f / fmaxf(deg[n], 1.0f);
  int c0 = l * 4;
  bf16* dst = e_f + (size_t)(E + n) * 128 + c0;
#pragma unroll
  for (int j = 0; j < 4; ++j)
    dst[j] = (bf16)(e_sum[(size_t)n * 128 + c0 + j] * inv);
}

// ---------------------------------------------------------------------------
// Kernel: pack fp32 (Ksrc x 128) weight into WMMA B-fragment layout, bf16
// element j (0..15) of lane l for tile (nb,kc): n = nb*16+(l&15),
// k = kc*32 + (l>>4)*8 + (j&7) + (j>>3)*16
// ---------------------------------------------------------------------------
__global__ void pack_b_kernel(const float* __restrict__ W, int Ksrc, int Kc,
                              bf16* __restrict__ dst) {
  int t = blockIdx.x * blockDim.x + threadIdx.x;
  int total = 8 * Kc * 512;
  if (t >= total) return;
  int j  = t & 15;
  int l  = (t >> 4) & 31;
  int rest = t >> 9;
  int kc = rest % Kc;
  int nb = rest / Kc;
  int n = nb * 16 + (l & 15);
  int k = kc * 32 + ((l >> 4) << 3) + (j & 7) + ((j >> 3) << 4);
  float v = (k < Ksrc) ? W[(size_t)k * 128 + n] : 0.0f;
  dst[t] = (bf16)v;
}

// ---------------------------------------------------------------------------
// Kernel: node transforms xl = h@Wl+bl, xr = h@Wr+br via WMMA
// one wave per (16-row group, 16-col tile nb)
// ---------------------------------------------------------------------------
__global__ void node_transform_kernel(const bf16* __restrict__ h_pad,
                                      const bf16* __restrict__ packWl,
                                      const bf16* __restrict__ packWr,
                                      const float* __restrict__ bl,
                                      const float* __restrict__ br,
                                      float* __restrict__ xl,
                                      float* __restrict__ xr, int N, int Kc) {
  int wid  = blockIdx.x * (blockDim.x >> 5) + (threadIdx.x >> 5);
  int lane = threadIdx.x & 31;
  int nb = wid & 7;
  int g  = wid >> 3;
  int row0 = g * 16;
  if (row0 >= N) return;
  v8f accL = {}; v8f accR = {};
  for (int kc = 0; kc < Kc; ++kc) {
    v16bf a  = load_a_frag(h_pad, row0, 128, kc, lane);
    v16bf bL = load_b_frag(packWl, nb, Kc, kc, lane);
    v16bf bR = load_b_frag(packWr, nb, Kc, kc, lane);
    accL = wmma_bf16(a, bL, accL);
    accR = wmma_bf16(a, bR, accR);
  }
  int col  = nb * 16 + (lane & 15);
  int half = lane >> 4;
  float vbl = bl[col], vbr = br[col];
#pragma unroll
  for (int r = 0; r < 8; ++r) {
    size_t idx = (size_t)(row0 + r + 8 * half) * 128 + col;
    xl[idx] = accL[r] + vbl;
    xr[idx] = accR[r] + vbr;
  }
}

// ---------------------------------------------------------------------------
// Kernel: per-layer init (out_accum = 0, node_max = -inf, node_sum = 0)
// ---------------------------------------------------------------------------
__global__ void init_layer_kernel(float* __restrict__ node_max,
                                  float* __restrict__ node_sum,
                                  float* __restrict__ out_accum, int N) {
  int t = blockIdx.x * blockDim.x + threadIdx.x;
  if (t < N * 4) { node_max[t] = -__builtin_inff(); node_sum[t] = 0.0f; }
  if (t < N * 128) out_accum[t] = 0.0f;
}

// ---------------------------------------------------------------------------
// Kernel: the big fused edge pass.
//   ee = e_f @ We (WMMA bf16, We staged in LDS via TDM), m = ee+xl[src]+xr[dst],
//   logits[e][h] = sum_c lrelu(m)*att ; atomic segment max per (dst, head)
// one wave per 16-edge group; 32 WMMAs per wave (8 N-tiles x 4 K-chunks)
// ---------------------------------------------------------------------------
__global__ void edge_logits_kernel(const bf16* __restrict__ e_f,
                                   const bf16* __restrict__ packWe,
                                   const float* __restrict__ xl,
                                   const float* __restrict__ xr,
                                   const int* __restrict__ src_idx,
                                   const int* __restrict__ dst_idx,
                                   const float* __restrict__ att,
                                   float* __restrict__ logits,
                                   float* __restrict__ node_max,
                                   int E, int total_rows) {
  __shared__ bf16 sWe[8 * 4 * 512];            // 32 KB packed B fragments
  stage_to_lds(packWe, sWe, 8 * 4 * 512 * (int)sizeof(bf16));

  int wid  = blockIdx.x * (blockDim.x >> 5) + (threadIdx.x >> 5);
  int lane = threadIdx.x & 31;
  int row0 = wid * 16;
  if (row0 >= total_rows) return;

  // prefetch next group's A rows (global_prefetch_b8)
  if (row0 + 16 < total_rows)
    __builtin_prefetch(e_f + (size_t)(row0 + 16 + (lane & 15)) * 128, 0, 1);

  v16bf a0 = load_a_frag(e_f, row0, 128, 0, lane);
  v16bf a1 = load_a_frag(e_f, row0, 128, 1, lane);
  v16bf a2 = load_a_frag(e_f, row0, 128, 2, lane);
  v16bf a3 = load_a_frag(e_f, row0, 128, 3, lane);

  int half = lane >> 4;
  int srcs[8], dsts[8];
#pragma unroll
  for (int r = 0; r < 8; ++r) {
    int e = row0 + r + 8 * half;
    if (e < E) { srcs[r] = src_idx[e]; dsts[r] = dst_idx[e]; }
    else       { srcs[r] = e - E;      dsts[r] = e - E; }
  }
  int nlane = lane & 15;

  for (int h = 0; h < 4; ++h) {        // head == nb>>1
    float part[8];
#pragma unroll
    for (int r = 0; r < 8; ++r) part[r] = 0.0f;
#pragma unroll
    for (int s = 0; s < 2; ++s) {
      int nb = h * 2 + s;
      v8f acc = {};
      acc = wmma_bf16(a0, load_b_frag(sWe, nb, 4, 0, lane), acc);
      acc = wmma_bf16(a1, load_b_frag(sWe, nb, 4, 1, lane), acc);
      acc = wmma_bf16(a2, load_b_frag(sWe, nb, 4, 2, lane), acc);
      acc = wmma_bf16(a3, load_b_frag(sWe, nb, 4, 3, lane), acc);
      int c = nb * 16 + nlane;
      float attc = att[c];
#pragma unroll
      for (int r = 0; r < 8; ++r) {
        float m = acc[r] + xl[(size_t)srcs[r] * 128 + c]
                         + xr[(size_t)dsts[r] * 128 + c];
        m = (m > 0.0f) ? m : m * NEG_SLOPE;
        part[r] += m * attc;
      }
    }
    // reduce over the 16 lanes holding the head's 32 channels
#pragma unroll
    for (int r = 0; r < 8; ++r) {
      float v = part[r];
      v += __shfl_xor(v, 1, 32);
      v += __shfl_xor(v, 2, 32);
      v += __shfl_xor(v, 4, 32);
      v += __shfl_xor(v, 8, 32);
      part[r] = v;
    }
    if (nlane == 0) {
#pragma unroll
      for (int r = 0; r < 8; ++r) {
        int e = row0 + r + 8 * half;
        logits[(size_t)e * 4 + h] = part[r];
        atomicMaxFloat(&node_max[dsts[r] * 4 + h], part[r]);
      }
    }
  }
}

// ---------------------------------------------------------------------------
// Kernel: ex = exp(logit - max[dst]); segment sum via atomics; ex in-place
// ---------------------------------------------------------------------------
__global__ void softmax_exp_kernel(float* __restrict__ logits,
                                   const float* __restrict__ node_max,
                                   float* __restrict__ node_sum,
                                   const int* __restrict__ dst_idx,
                                   int E, int total_rows) {
  int e = blockIdx.x * blockDim.x + threadIdx.x;
  if (e >= total_rows) return;
  int d = (e < E) ? dst_idx[e] : (e - E);
#pragma unroll
  for (int h = 0; h < 4; ++h) {
    float ex = __expf(logits[(size_t)e * 4 + h] - node_max[d * 4 + h]);
    logits[(size_t)e * 4 + h] = ex;
    atomicAdd(&node_sum[d * 4 + h], ex);
  }
}

// ---------------------------------------------------------------------------
// Kernel: out[dst] += xl[src] * alpha  (L2-resident float atomics)
// one wave per edge row; lane handles 4 channels (single head per lane)
// ---------------------------------------------------------------------------
__global__ void scatter_kernel(const float* __restrict__ ex,
                               const float* __restrict__ node_sum,
                               const float* __restrict__ xl,
                               const int* __restrict__ src_idx,
                               const int* __restrict__ dst_idx,
                               float* __restrict__ out_accum,
                               int E, int total_rows) {
  int t = blockIdx.x * blockDim.x + threadIdx.x;
  int e = t >> 5, l = t & 31;
  if (e >= total_rows) return;
  int s = (e < E) ? src_idx[e] : (e - E);
  int d = (e < E) ? dst_idx[e] : (e - E);
  int h = l >> 3;
  float alpha = ex[(size_t)e * 4 + h] / node_sum[d * 4 + h];
  int c0 = l * 4;
  const float* xs = xl + (size_t)s * 128 + c0;
  float* od = out_accum + (size_t)d * 128 + c0;
#pragma unroll
  for (int j = 0; j < 4; ++j) atomicAdd(od + j, xs[j] * alpha);
}

// ---------------------------------------------------------------------------
// Kernel: +bias, LayerNorm, SiLU, residual -> h f32 and h_pad bf16
// one wave per node row
// ---------------------------------------------------------------------------
__global__ void finalize_kernel(const float* __restrict__ out_accum,
                                const float* __restrict__ bias,
                                const float* __restrict__ ln_g,
                                const float* __restrict__ ln_b,
                                const float* __restrict__ h_in,
                                float* __restrict__ h_out,
                                bf16* __restrict__ h_pad,
                                int N, int do_residual) {
  int wid  = blockIdx.x * (blockDim.x >> 5) + (threadIdx.x >> 5);
  int lane = threadIdx.x & 31;
  if (wid >= N) return;
  const float* row = out_accum + (size_t)wid * 128;
  float v[4];
  float s = 0.0f;
#pragma unroll
  for (int i = 0; i < 4; ++i) {
    v[i] = row[lane + 32 * i] + bias[lane + 32 * i];
    s += v[i];
  }
#pragma unroll
  for (int m = 1; m < 32; m <<= 1) s += __shfl_xor(s, m, 32);
  float mean = s * (1.0f / 128.0f);
  float q = 0.0f;
#pragma unroll
  for (int i = 0; i < 4; ++i) { float d = v[i] - mean; q += d * d; }
#pragma unroll
  for (int m = 1; m < 32; m <<= 1) q += __shfl_xor(q, m, 32);
  float rstd = rsqrtf(q * (1.0f / 128.0f) + 1e-5f);
#pragma unroll
  for (int i = 0; i < 4; ++i) {
    int c = lane + 32 * i;
    float hn = (v[i] - mean) * rstd * ln_g[c] + ln_b[c];
    hn = silu(hn);
    float out = do_residual ? (h_in[(size_t)wid * 128 + c] + hn) : hn;
    h_out[(size_t)wid * 128 + c] = out;
    h_pad[(size_t)wid * 128 + c] = (bf16)out;
  }
}

// ---------------------------------------------------------------------------
// Host orchestration
// ---------------------------------------------------------------------------
extern "C" void kernel_launch(void* const* d_in, const int* in_sizes, int n_in,
                              void* d_out, int out_size, void* d_ws, size_t ws_size,
                              hipStream_t stream) {
  const float* x         = (const float*)d_in[0];
  const float* edge_attr = (const float*)d_in[1];
  const float* edge_w    = (const float*)d_in[2];
  const float* edge_b    = (const float*)d_in[3];
  const int*   edge_idx  = (const int*)d_in[n_in - 1];

  const int N = in_sizes[0] / 12;
  const int E = in_sizes[1] / 9;
  const int T = E + N;                 // total rows incl. self-loops
  const int* src_idx = edge_idx;
  const int* dst_idx = edge_idx + E;

  // ---- workspace carve-up (256B aligned) ----
  char* w = (char*)d_ws;
  auto alloc = [&](size_t bytes) -> void* {
    void* p = (void*)w;
    w += (bytes + 255) & ~(size_t)255;
    return p;
  };
  float* h      = (float*)alloc((size_t)N * 128 * 4);
  bf16*  h_pad  = (bf16*) alloc((size_t)N * 128 * 2);
  bf16*  e_f    = (bf16*) alloc((size_t)T * 128 * 2);
  float* e_sum  = (float*)alloc((size_t)N * 128 * 4);   // reused as out_accum
  float* deg    = (float*)alloc((size_t)N * 4);
  float* xl     = (float*)alloc((size_t)N * 128 * 4);
  float* xr     = (float*)alloc((size_t)N * 128 * 4);
  float* logits = (float*)alloc((size_t)T * 4 * 4);
  float* nmax   = (float*)alloc((size_t)N * 4 * 4);
  float* nsum   = (float*)alloc((size_t)N * 4 * 4);
  bf16*  packWl = (bf16*) alloc(8 * 4 * 512 * 2);
  bf16*  packWr = (bf16*) alloc(8 * 4 * 512 * 2);
  bf16*  packWe = (bf16*) alloc(8 * 4 * 512 * 2);
  float* out_accum = e_sum;

  const int B = 256;

  // 1) positional encoding -> h_pad
  pe_kernel<<<cdiv(N, B), B, 0, stream>>>(x, h_pad, N);

  // 2) edge MLP (+ deg / e_sum accumulation)
  zero_kernel<<<cdiv((long long)N * 128, B), B, 0, stream>>>(e_sum, N * 128);
  zero_kernel<<<cdiv(N, B), B, 0, stream>>>(deg, N);
  edge_mlp_kernel<<<cdiv((long long)E * 32, B), B, 0, stream>>>(
      edge_attr, edge_w, edge_b, dst_idx, e_f, e_sum, deg, E);

  // 3) self-loop mean features
  emean_kernel<<<cdiv((long long)N * 32, B), B, 0, stream>>>(e_sum, deg, e_f, N, E);

  // 4) layers
  for (int li = 0; li < 4; ++li) {
    int base = 4 + 9 * li;
    const float* Wl   = (const float*)d_in[base + 0];
    const float* bl   = (const float*)d_in[base + 1];
    const float* Wr   = (const float*)d_in[base + 2];
    const float* br   = (const float*)d_in[base + 3];
    const float* We   = (const float*)d_in[base + 4];
    const float* att  = (const float*)d_in[base + 5];
    const float* bias = (const float*)d_in[base + 6];
    const float* ln_g = (const float*)d_in[base + 7];
    const float* ln_b = (const float*)d_in[base + 8];
    int Ksrc = in_sizes[base + 0] / 128;          // 44 or 128
    int Kc   = (Ksrc + 31) / 32;                  // 2 or 4 K-chunks

    pack_b_kernel<<<cdiv(8 * Kc * 512, B), B, 0, stream>>>(Wl, Ksrc, Kc, packWl);
    pack_b_kernel<<<cdiv(8 * Kc * 512, B), B, 0, stream>>>(Wr, Ksrc, Kc, packWr);
    pack_b_kernel<<<cdiv(8 * 4 * 512, B), B, 0, stream>>>(We, 128, 4, packWe);

    // xl / xr node transforms (WMMA): cdiv(N,16)*8 waves
    {
      long long waves = (long long)cdiv(N, 16) * 8;
      node_transform_kernel<<<cdiv(waves * 32, B), B, 0, stream>>>(
          h_pad, packWl, packWr, bl, br, xl, xr, N, Kc);
    }

    init_layer_kernel<<<cdiv((long long)N * 128, B), B, 0, stream>>>(nmax, nsum,
                                                                     out_accum, N);

    // fused ee-GEMM + attention logits + segment max (We staged via TDM->LDS)
    {
      long long waves = cdiv(T, 16);
      edge_logits_kernel<<<cdiv(waves * 32, B), B, 0, stream>>>(
          e_f, packWe, xl, xr, src_idx, dst_idx, att, logits, nmax, E, T);
    }

    softmax_exp_kernel<<<cdiv(T, B), B, 0, stream>>>(logits, nmax, nsum,
                                                     dst_idx, E, T);

    scatter_kernel<<<cdiv((long long)T * 32, B), B, 0, stream>>>(
        logits, nsum, xl, src_idx, dst_idx, out_accum, E, T);

    float* hout = (li == 3) ? (float*)d_out : h;
    finalize_kernel<<<cdiv((long long)N * 32, B), B, 0, stream>>>(
        out_accum, bias, ln_g, ln_b, h, hout, h_pad, N, li > 0 ? 1 : 0);
  }
}
